// LMGNN_12816182411897
// MI455X (gfx1250) — matrix operand order repaired and checked by reading, not compiled, not run on hardware
//
#include <hip/hip_runtime.h>
#include <math.h>

#define NNODES   24000
#define DM       128
#define DI       256      // D_INNER
#define DS       16       // D_STATE
#define DTR      8        // DT_RANK
#define DBC_LD   40       // dt_rank + 2*d_state
#define CHUNK    64
#define NCHUNK   (NNODES / CHUNK)   // 375, exact

typedef __bf16 bf16;
typedef __attribute__((ext_vector_type(16))) __bf16 v16bf;
typedef __attribute__((ext_vector_type(8)))  float  v8f;
typedef __attribute__((ext_vector_type(4)))  unsigned v4u;
typedef __attribute__((ext_vector_type(8)))  int      v8i;
typedef __attribute__((ext_vector_type(4)))  int      v4i;

// ---------------------------------------------------------------- utilities
__global__ void zero_f32(float* __restrict__ p, int n) {
    int i = blockIdx.x * blockDim.x + threadIdx.x;
    if (i < n) p[i] = 0.0f;
}

__global__ void f32_to_bf16(const float* __restrict__ s, bf16* __restrict__ d, int n) {
    int i = blockIdx.x * blockDim.x + threadIdx.x;
    if (i < n) d[i] = (bf16)s[i];
}

// ---------------------------------------------------------------- SpMM: one edge per wave32, float4 per lane
__global__ void spmm_scatter(const int* __restrict__ rows, const int* __restrict__ cols,
                             const float* __restrict__ vals,
                             const float* __restrict__ ein, float* __restrict__ eout, int nnz) {
    int idx  = blockIdx.x * blockDim.x + threadIdx.x;
    int edge = idx >> 5;
    if (edge >= nnz) return;
    int q = idx & 31;
    int r = rows[edge], c = cols[edge];
    float v = vals[edge];
    float4 m = ((const float4*)(ein + (size_t)c * DM))[q];
    float* dst = eout + (size_t)r * DM + q * 4;
    atomicAdd(dst + 0, v * m.x);
    atomicAdd(dst + 1, v * m.y);
    atomicAdd(dst + 2, v * m.z);
    atomicAdd(dst + 3, v * m.w);
}

// ---------------------------------------------------------------- row L2 norm (wave32 per row)
__global__ void l2norm_rows(const float* __restrict__ in, float* __restrict__ out, int nrows) {
    int row = blockIdx.x * 8 + (threadIdx.x >> 5);
    if (row >= nrows) return;
    int lane = threadIdx.x & 31;
    const float* x = in + (size_t)row * DM;
    float v0 = x[lane], v1 = x[lane + 32], v2 = x[lane + 64], v3 = x[lane + 96];
    float s = v0 * v0 + v1 * v1 + v2 * v2 + v3 * v3;
    #pragma unroll
    for (int off = 16; off; off >>= 1) s += __shfl_xor(s, off, 32);
    float inv = 1.0f / fmaxf(sqrtf(s), 1e-12f);
    float* y = out + (size_t)row * DM;
    y[lane] = v0 * inv; y[lane + 32] = v1 * inv; y[lane + 64] = v2 * inv; y[lane + 96] = v3 * inv;
}

// ---------------------------------------------------------------- residual add + LayerNorm (wave32 per row)
__global__ void add_layernorm(const float* __restrict__ X, const float* __restrict__ R,
                              const float* __restrict__ g, const float* __restrict__ b,
                              float* __restrict__ out, int nrows) {
    int row = blockIdx.x * 8 + (threadIdx.x >> 5);
    if (row >= nrows) return;
    int lane = threadIdx.x & 31;
    const float* xp = X + (size_t)row * DM;
    const float* rp = R + (size_t)row * DM;
    float v[4];
    #pragma unroll
    for (int i = 0; i < 4; ++i) v[i] = xp[lane + 32 * i] + rp[lane + 32 * i];
    float s = v[0] + v[1] + v[2] + v[3];
    #pragma unroll
    for (int off = 16; off; off >>= 1) s += __shfl_xor(s, off, 32);
    float mu = s * (1.0f / 128.0f);
    float q = 0.0f;
    #pragma unroll
    for (int i = 0; i < 4; ++i) { float d = v[i] - mu; q += d * d; }
    #pragma unroll
    for (int off = 16; off; off >>= 1) q += __shfl_xor(q, off, 32);
    float rs = rsqrtf(q * (1.0f / 128.0f) + 1e-12f);
    float* yp = out + (size_t)row * DM;
    #pragma unroll
    for (int i = 0; i < 4; ++i) {
        int cc = lane + 32 * i;
        yp[cc] = (v[i] - mu) * rs * g[cc] + b[cc];
    }
}

// ---------------------------------------------------------------- TDM 2D tile load (D# built per ISA ch.8)
__device__ __forceinline__ unsigned lds_off(const void* p) {
    return (unsigned)(unsigned long long)p;   // low 32 bits of flat LDS addr = LDS byte offset
}

// g1w0 = data_size | pad config word (group1 word0)
#define G1W0_A ((2u << 16) | (1u << 20) | (4u << 22))   // 4B elems, pad 1 DW per 32 DW
#define G1W0_W ((1u << 16) | (1u << 20) | (3u << 22))   // 2B elems, pad 1 DW per 16 DW

__device__ __forceinline__ void tdm_load_2d(unsigned ldsaddr, const void* gaddr,
                                            unsigned g1w0, unsigned tdim0, unsigned tdim1,
                                            unsigned tile0, unsigned tile1,
                                            unsigned long long stride0) {
    unsigned long long ga = (unsigned long long)gaddr;
    v4u g0;
    g0[0] = 1u;                                              // count=1, user descriptor
    g0[1] = ldsaddr;                                         // lds_addr (bytes)
    g0[2] = (unsigned)(ga & 0xffffffffu);                    // global_addr[31:0]
    g0[3] = (unsigned)((ga >> 32) & 0x1ffffffu) | (2u << 30); // global_addr[56:32] | type=2
    v8i g1;
    g1[0] = (int)g1w0;                                       // mask=0, data_size, pad cfg
    g1[1] = (int)((tdim0 & 0xffffu) << 16);                  // tensor_dim0[15:0]
    g1[2] = (int)((tdim0 >> 16) | ((tdim1 & 0xffffu) << 16));// tensor_dim0[31:16] | tensor_dim1[15:0]
    g1[3] = (int)((tdim1 >> 16) | (tile0 << 16));            // tensor_dim1[31:16] | tile_dim0
    g1[4] = (int)(tile1 & 0xffffu);                          // tile_dim1, tile_dim2=0 (2D)
    g1[5] = (int)(unsigned)(stride0 & 0xffffffffu);          // tensor_dim0_stride[31:0]
    g1[6] = (int)(unsigned)((stride0 >> 32) & 0xffffu);      // stride[47:32], dim1_stride lo=0
    g1[7] = 0;
    v4i z4 = {0, 0, 0, 0};
    v8i z8 = {0, 0, 0, 0, 0, 0, 0, 0};
    __builtin_amdgcn_tensor_load_to_lds(g0, g1, z4, z4, z8, 0);
}

// ---------------------------------------------------------------- WMMA GEMM: C[M,N] = A[M,K](f32) * W[N,K](bf16)^T
// mode: 0=plain, 1=+bias, 2=+bias then leaky_relu(0.01)
#define BM 128
#define BN 64
#define ADW 33    // A LDS row pitch in DWORDs (32 + TDM pad) -> bank-conflict free
#define WPITCH 34 // W LDS row pitch in bf16 (17 DWORDs)

__global__ __launch_bounds__(256)
void gemm_bf16_wmma(const float* __restrict__ A, const bf16* __restrict__ W,
                    const float* __restrict__ bias, float* __restrict__ C,
                    int M, int K, int N, int ldc, int mode) {
    __shared__ float AsBuf[2][BM * ADW];
    __shared__ bf16  WsBuf[2][BN * WPITCH];
    const int tid  = threadIdx.x;
    const int lane = tid & 31;
    const int wv   = tid >> 5;          // 8 waves: wave owns 16 M-rows x 64 N-cols
    const int half = lane >> 4;
    const int l16  = lane & 15;
    const int m0   = blockIdx.x * BM;
    const int n0   = blockIdx.y * BN;
    const int KT   = K >> 5;

    v8f acc0 = {}, acc1 = {}, acc2 = {}, acc3 = {};

    const unsigned aLds0 = lds_off(&AsBuf[0][0]), aLds1 = lds_off(&AsBuf[1][0]);
    const unsigned wLds0 = lds_off(&WsBuf[0][0]), wLds1 = lds_off(&WsBuf[1][0]);
    const unsigned tdimA = (unsigned)(M - m0);   // rows remaining: OOB rows zero-filled by TDM
    const unsigned tdimW = (unsigned)(N > n0 ? N - n0 : 0);

    if (wv == 0) {
        tdm_load_2d(aLds0, A + (size_t)m0 * K, G1W0_A, 32u, tdimA, 32u, BM, (unsigned long long)K);
        tdm_load_2d(wLds0, W + (size_t)n0 * K, G1W0_W, 32u, tdimW, 32u, BN, (unsigned long long)K);
    }

    for (int kt = 0; kt < KT; ++kt) {
        const int cur = kt & 1;
        if (wv == 0) {
            if (kt + 1 < KT) {
                tdm_load_2d(cur ? aLds0 : aLds1, A + (size_t)m0 * K + (kt + 1) * 32,
                            G1W0_A, 32u, tdimA, 32u, BM, (unsigned long long)K);
                tdm_load_2d(cur ? wLds0 : wLds1, W + (size_t)n0 * K + (kt + 1) * 32,
                            G1W0_W, 32u, tdimW, 32u, BN, (unsigned long long)K);
                __builtin_amdgcn_s_wait_tensorcnt(2);   // current pair done; next pair in flight
            } else {
                __builtin_amdgcn_s_wait_tensorcnt(0);
            }
        }
        __syncthreads();

        const float* As = AsBuf[cur];
        const bf16*  Ws = WsBuf[cur];

        // A fragment per ISA 7.12.2: lanes 0-15 -> K {0..7,16..23}, lanes 16-31 -> K {8..15,24..31}
        v16bf af;
        int m = wv * 16 + l16;
        #pragma unroll
        for (int v = 0; v < 8; ++v) {
            int k = ((v < 4) ? (2 * v) : (16 + 2 * (v - 4))) + (half ? 8 : 0);
            af[2 * v]     = (bf16)As[m * ADW + k];
            af[2 * v + 1] = (bf16)As[m * ADW + k + 1];
        }
        // B fragments: lane = column N, same K order as A: lanes 0-15 -> K {0..7,16..23}, lanes 16-31 -> K {8..15,24..31}
        #pragma unroll
        for (int nb = 0; nb < 4; ++nb) {
            v16bf bfr;
            int n = nb * 16 + l16;
            #pragma unroll
            for (int v = 0; v < 8; ++v) {
                int k = ((v < 4) ? (2 * v) : (16 + 2 * (v - 4))) + (half ? 8 : 0);
                bfr[2 * v]     = Ws[n * WPITCH + k];
                bfr[2 * v + 1] = Ws[n * WPITCH + k + 1];
            }
            v8f* ap = (nb == 0) ? &acc0 : (nb == 1) ? &acc1 : (nb == 2) ? &acc2 : &acc3;
            *ap = __builtin_amdgcn_wmma_f32_16x16x32_bf16(false, af, false, bfr,
                                                          (short)0, *ap, false, false);
        }
        __syncthreads();
    }

    #pragma unroll
    for (int nb = 0; nb < 4; ++nb) {
        v8f acc = (nb == 0) ? acc0 : (nb == 1) ? acc1 : (nb == 2) ? acc2 : acc3;
        int gn = n0 + nb * 16 + l16;
        if (gn >= N) continue;
        float bv = (mode > 0 && bias) ? bias[gn] : 0.0f;
        #pragma unroll
        for (int v = 0; v < 8; ++v) {
            int gm = m0 + wv * 16 + v + 8 * half;   // C/D layout: M = v + 8*half
            if (gm >= M) continue;
            float x = acc[v] + bv;
            if (mode == 2) x = (x < 0.0f) ? 0.01f * x : x;
            C[(size_t)gm * ldc + gn] = x;
        }
    }
}

// ---------------------------------------------------------------- depthwise causal conv(4) + SiLU
__global__ void conv_silu(const float* __restrict__ xi, const float* __restrict__ cw,
                          const float* __restrict__ cb, float* __restrict__ u, int L) {
    int idx = blockIdx.x * blockDim.x + threadIdx.x;
    if (idx >= L * DI) return;
    int d = idx & (DI - 1);
    int t = idx >> 8;
    float acc = cb[d];
    #pragma unroll
    for (int k = 0; k < 4; ++k) {
        int tt = t + k - 3;
        float xv = (tt >= 0) ? xi[(size_t)tt * DI + d] : 0.0f;
        acc += xv * cw[d * 4 + k];
    }
    u[idx] = acc / (1.0f + expf(-acc));
}

// ---------------------------------------------------------------- dt = softplus(dbc[:, :8] @ dt_w^T + dt_b)
__global__ void dt_proj_softplus(const float* __restrict__ dbc, const float* __restrict__ dtw,
                                 const float* __restrict__ dtb, float* __restrict__ dt, int L) {
    int idx = blockIdx.x * blockDim.x + threadIdx.x;
    if (idx >= L * DI) return;
    int d = idx & (DI - 1);
    int t = idx >> 8;
    const float* row = dbc + (size_t)t * DBC_LD;
    float s = dtb[d];
    #pragma unroll
    for (int r = 0; r < DTR; ++r) s += row[r] * dtw[d * DTR + r];
    dt[idx] = (s > 20.0f) ? s : log1pf(expf(s));
}

// ---------------------------------------------------------------- chunked selective scan, pass 1
__global__ __launch_bounds__(256)
void scan_pass1(const float* __restrict__ u, const float* __restrict__ dtv,
                const float* __restrict__ dbc, const float* __restrict__ A_log,
                float* __restrict__ Aprod, float* __restrict__ Hpart, int L) {
    __shared__ float sB[CHUNK][DS];
    int c  = blockIdx.x;
    int t0 = c * CHUNK;
    int nt = min(CHUNK, L - t0);
    int tid = threadIdx.x;
    for (int i = tid; i < nt * DS; i += 256) {
        int tt = i >> 4, j = i & 15;
        sB[tt][j] = dbc[(size_t)(t0 + tt) * DBC_LD + DTR + j];
    }
    __syncthreads();
    int d = tid;
    float Av[DS];
    #pragma unroll
    for (int s = 0; s < DS; ++s) Av[s] = -expf(A_log[d * DS + s]);
    float h[DS];
    #pragma unroll
    for (int s = 0; s < DS; ++s) h[s] = 0.0f;
    float dtsum = 0.0f;
    for (int tt = 0; tt < nt; ++tt) {
        int t = t0 + tt;
        float dtt = dtv[(size_t)t * DI + d];
        float du  = dtt * u[(size_t)t * DI + d];
        dtsum += dtt;
        #pragma unroll
        for (int s = 0; s < DS; ++s) {
            float a = expf(dtt * Av[s]);
            h[s] = a * h[s] + du * sB[tt][s];
        }
    }
    size_t base = ((size_t)c * DI + d) * DS;
    #pragma unroll
    for (int s = 0; s < DS; ++s) {
        Aprod[base + s] = expf(dtsum * Av[s]);   // prod exp(dt*A) == exp(A*sum dt)
        Hpart[base + s] = h[s];
    }
}

// ---------------------------------------------------------------- pass 2: chain 4096 recurrences across chunks
__global__ void scan_pass2(const float* __restrict__ Aprod, const float* __restrict__ Hpart,
                           float* __restrict__ Hinit, int nch) {
    int i = blockIdx.x * blockDim.x + threadIdx.x;
    if (i >= DI * DS) return;
    float h = 0.0f;
    for (int c = 0; c < nch; ++c) {
        size_t o = (size_t)c * (DI * DS) + i;
        Hinit[o] = h;
        h = Aprod[o] * h + Hpart[o];
    }
}

// ---------------------------------------------------------------- pass 3: replay with true init, fuse y/gate
__global__ __launch_bounds__(256)
void scan_pass3(const float* __restrict__ u, const float* __restrict__ dtv,
                const float* __restrict__ dbc, const float* __restrict__ A_log,
                const float* __restrict__ Hinit, const float* __restrict__ z,
                const float* __restrict__ Dp, float* __restrict__ ys, int L) {
    __shared__ float sB[CHUNK][DS];
    __shared__ float sC[CHUNK][DS];
    int c  = blockIdx.x;
    int t0 = c * CHUNK;
    int nt = min(CHUNK, L - t0);
    int tid = threadIdx.x;
    for (int i = tid; i < nt * 32; i += 256) {
        int tt = i >> 5, j = i & 31;
        float v = dbc[(size_t)(t0 + tt) * DBC_LD + DTR + j];
        if (j < DS) sB[tt][j] = v; else sC[tt][j - DS] = v;
    }
    __syncthreads();
    int d = tid;
    float Av[DS];
    #pragma unroll
    for (int s = 0; s < DS; ++s) Av[s] = -expf(A_log[d * DS + s]);
    float h[DS];
    size_t base = ((size_t)c * DI + d) * DS;
    #pragma unroll
    for (int s = 0; s < DS; ++s) h[s] = Hinit[base + s];
    float dval = Dp[d];
    for (int tt = 0; tt < nt; ++tt) {
        int t = t0 + tt;
        float dtt = dtv[(size_t)t * DI + d];
        float uv  = u[(size_t)t * DI + d];
        float du  = dtt * uv;
        float y = 0.0f;
        #pragma unroll
        for (int s = 0; s < DS; ++s) {
            float a = expf(dtt * Av[s]);
            h[s] = a * h[s] + du * sB[tt][s];
            y += h[s] * sC[tt][s];
        }
        float zv = z[(size_t)t * DI + d];
        float gate = zv / (1.0f + expf(-zv));
        ys[(size_t)t * DI + d] = (y + uv * dval) * gate;
    }
}

// ================================================================ host orchestration
extern "C" void kernel_launch(void* const* d_in, const int* in_sizes, int n_in,
                              void* d_out, int out_size, void* d_ws, size_t ws_size,
                              hipStream_t stream) {
    (void)n_in; (void)out_size; (void)ws_size;
    const float* user_emb = (const float*)d_in[0];
    const float* item_emb = (const float*)d_in[1];
    const int*   adj_rows = (const int*)d_in[2];
    const int*   adj_cols = (const int*)d_in[3];
    const float* adj_vals = (const float*)d_in[4];
    const int nnz = in_sizes[2];
    const int N = NNODES;

    char* ws = (char*)d_ws;
    size_t off = 0;
    auto alloc = [&](size_t bytes) -> void* {
        off = (off + 255) & ~((size_t)255);
        void* p = ws + off;
        off += bytes;
        return p;
    };
    float* EA    = (float*)alloc((size_t)N * DM * 4);
    float* EB    = (float*)alloc((size_t)N * DM * 4);
    float* XI    = (float*)alloc((size_t)N * DI * 4);   // also reused as ys
    float* Zb    = (float*)alloc((size_t)N * DI * 4);
    float* U     = (float*)alloc((size_t)N * DI * 4);   // also reused as FF inner
    float* DBC   = (float*)alloc((size_t)N * DBC_LD * 4);
    float* DT    = (float*)alloc((size_t)N * DI * 4);
    float* APROD = (float*)alloc((size_t)NCHUNK * DI * DS * 4);
    float* HPART = (float*)alloc((size_t)NCHUNK * DI * DS * 4);
    float* HINIT = (float*)alloc((size_t)NCHUNK * DI * DS * 4);
    bf16* Wb_in  = (bf16*)alloc((size_t)512 * 128 * 2);
    bf16* Wb_xp  = (bf16*)alloc((size_t)40 * 256 * 2);
    bf16* Wb_out = (bf16*)alloc((size_t)128 * 256 * 2);
    bf16* Wb_w1  = (bf16*)alloc((size_t)256 * 128 * 2);
    bf16* Wb_w2  = (bf16*)alloc((size_t)128 * 256 * 2);

    auto gemm = [&](const float* A, const bf16* W, const float* bias, float* C,
                    int M, int K, int Nn, int ldc, int mode) {
        dim3 g((M + BM - 1) / BM, (Nn + BN - 1) / BN);
        gemm_bf16_wmma<<<g, 256, 0, stream>>>(A, W, bias, C, M, K, Nn, ldc, mode);
    };

    // ---- concat embeddings
    (void)hipMemcpyAsync(EA, user_emb, (size_t)10000 * DM * 4, hipMemcpyDeviceToDevice, stream);
    (void)hipMemcpyAsync(EA + (size_t)10000 * DM, item_emb, (size_t)14000 * DM * 4,
                         hipMemcpyDeviceToDevice, stream);

    // ---- 3x GNN propagation
    for (int it = 0; it < 3; ++it) {
        zero_f32<<<(N * DM + 255) / 256, 256, 0, stream>>>(EB, N * DM);
        spmm_scatter<<<(nnz * 32 + 255) / 256, 256, 0, stream>>>(adj_rows, adj_cols, adj_vals,
                                                                 EA, EB, nnz);
        l2norm_rows<<<(N + 7) / 8, 256, 0, stream>>>(EB, EA, N);
    }

    // ---- 2x Mamba layers
    for (int layer = 0; layer < 2; ++layer) {
        int b = 5 + 17 * layer;
        const float* in_w   = (const float*)d_in[b + 0];
        const float* conv_w = (const float*)d_in[b + 1];
        const float* conv_b = (const float*)d_in[b + 2];
        const float* xproj  = (const float*)d_in[b + 3];
        const float* dt_w   = (const float*)d_in[b + 4];
        const float* dt_b   = (const float*)d_in[b + 5];
        const float* A_log  = (const float*)d_in[b + 6];
        const float* Dp     = (const float*)d_in[b + 7];
        const float* out_w  = (const float*)d_in[b + 8];
        const float* ln1g   = (const float*)d_in[b + 9];
        const float* ln1b   = (const float*)d_in[b + 10];
        const float* w1     = (const float*)d_in[b + 11];
        const float* b1     = (const float*)d_in[b + 12];
        const float* w2     = (const float*)d_in[b + 13];
        const float* b2     = (const float*)d_in[b + 14];
        const float* ln2g   = (const float*)d_in[b + 15];
        const float* ln2b   = (const float*)d_in[b + 16];

        f32_to_bf16<<<(512 * 128 + 255) / 256, 256, 0, stream>>>(in_w, Wb_in, 512 * 128);
        f32_to_bf16<<<(40 * 256 + 255) / 256, 256, 0, stream>>>(xproj, Wb_xp, 40 * 256);
        f32_to_bf16<<<(128 * 256 + 255) / 256, 256, 0, stream>>>(out_w, Wb_out, 128 * 256);
        f32_to_bf16<<<(256 * 128 + 255) / 256, 256, 0, stream>>>(w1, Wb_w1, 256 * 128);
        f32_to_bf16<<<(128 * 256 + 255) / 256, 256, 0, stream>>>(w2, Wb_w2, 128 * 256);

        // in-proj: xi = x @ in_w[:256].T , z = x @ in_w[256:].T
        gemm(EA, Wb_in,             nullptr, XI, N, 128, 256, 256, 0);
        gemm(EA, Wb_in + 256 * 128, nullptr, Zb, N, 128, 256, 256, 0);
        conv_silu<<<(N * DI + 255) / 256, 256, 0, stream>>>(XI, conv_w, conv_b, U, N);
        gemm(U, Wb_xp, nullptr, DBC, N, 256, 40, DBC_LD, 0);
        dt_proj_softplus<<<(N * DI + 255) / 256, 256, 0, stream>>>(DBC, dt_w, dt_b, DT, N);
        scan_pass1<<<NCHUNK, 256, 0, stream>>>(U, DT, DBC, A_log, APROD, HPART, N);
        scan_pass2<<<16, 256, 0, stream>>>(APROD, HPART, HINIT, NCHUNK);
        scan_pass3<<<NCHUNK, 256, 0, stream>>>(U, DT, DBC, A_log, HINIT, Zb, Dp, XI, N);
        // out-proj + residual LN
        gemm(XI, Wb_out, nullptr, EB, N, 256, 128, 128, 0);
        add_layernorm<<<(N + 7) / 8, 256, 0, stream>>>(EB, EA, ln1g, ln1b, EA, N);
        // FFN
        gemm(EA, Wb_w1, b1, U, N, 128, 256, 256, 2);
        gemm(U, Wb_w2, b2, EB, N, 256, 128, 128, 1);
        add_layernorm<<<(N + 7) / 8, 256, 0, stream>>>(EB, EA, ln2g, ln2b, EA, N);
    }

    (void)hipMemcpyAsync(d_out, EA, (size_t)N * DM * 4, hipMemcpyDeviceToDevice, stream);
}
